// TemporalCrossAttent_61349312856384
// MI455X (gfx1250) — matrix operand order repaired and checked by reading, not compiled, not run on hardware
//
#include <hip/hip_runtime.h>

// TemporalCrossAttent fused kernel for MI455X (gfx1250, wave32, WMMA).
// One wave per batch; all intermediates (k, v^T, q, sim, attn) live in LDS.
// feat2 A-panel is loaded ONCE into VGPRs and reused for both Wk and Wv
// projections -> HBM read traffic is exactly feat1 + feat2 (+ L2-resident W).

#define BATCHES 16384
#define TSEQ    20
#define CIN     256
#define HIDN    256

typedef _Float16 v16h __attribute__((ext_vector_type(16)));
typedef _Float16 v8h  __attribute__((ext_vector_type(8)));
typedef float    v8f  __attribute__((ext_vector_type(8)));

#define WAVES_PER_BLOCK 4
// Per-wave LDS slot layout (bytes):
#define K_OFF   0        // k   : [32][256] f16  (rows >=20 hold clamped dup of row 19; masked later)
#define VT_OFF  16384    // v^T : [256][32] f16  (cols >=20 never multiplied by nonzero attn)
#define Q_OFF   32768    // q   : [32][256] f16
#define SIM_OFF 49152    // sim : [32][32]  f32
#define ATT_OFF 53248    // attn: [32][32]  f16
#define SLOT_BYTES 55296
#define LDS_BYTES (SLOT_BYTES * WAVES_PER_BLOCK)   // 216 KB of the 320 KB WGP pool

#define WMMA_F16(A, B, C) \
  __builtin_amdgcn_wmma_f32_16x16x32_f16(false, (A), false, (B), (short)0, (C), false, false)

#define LDS_FENCE() asm volatile("s_wait_dscnt 0" ::: "memory")

// ---- A-operand (16x32 f16) from global fp32 rows.
// Layout (16-bit A 16x32): lanes 0-15 = rows, K runs {c0..c0+7, c0+16..c0+23};
// lanes 16-31 = same rows, K runs {c0+8..c0+15, c0+24..c0+31}.
static __device__ inline v16h a_from_global(const float* row, int c0, bool hi) {
  const float* p = row + c0 + (hi ? 8 : 0);
  float4 x0 = *(const float4*)(p + 0);
  float4 x1 = *(const float4*)(p + 4);
  float4 y0 = *(const float4*)(p + 16);
  float4 y1 = *(const float4*)(p + 20);
  v16h a;
  a[0]=(_Float16)x0.x;  a[1]=(_Float16)x0.y;  a[2]=(_Float16)x0.z;  a[3]=(_Float16)x0.w;
  a[4]=(_Float16)x1.x;  a[5]=(_Float16)x1.y;  a[6]=(_Float16)x1.z;  a[7]=(_Float16)x1.w;
  a[8]=(_Float16)y0.x;  a[9]=(_Float16)y0.y;  a[10]=(_Float16)y0.z; a[11]=(_Float16)y0.w;
  a[12]=(_Float16)y1.x; a[13]=(_Float16)y1.y; a[14]=(_Float16)y1.z; a[15]=(_Float16)y1.w;
  return a;
}

// ---- B-operand (32x16 f16) from global fp32 weights W[h][c] (B[c][h] = W[h][c]).
// Layout: lane n<16 = column h, K = {c0..c0+7, c0+16..c0+23}; lane n>=16 = same column, K = {c0+8..c0+15, c0+24..c0+31}.
static __device__ inline v16h b_from_global(const float* W, int h, int c0, bool hi) {
  const float* p = W + h * CIN + c0 + (hi ? 8 : 0);
  float4 x0 = *(const float4*)(p + 0);
  float4 x1 = *(const float4*)(p + 4);
  float4 x2 = *(const float4*)(p + 16);
  float4 x3 = *(const float4*)(p + 20);
  v16h b;
  b[0]=(_Float16)x0.x;  b[1]=(_Float16)x0.y;  b[2]=(_Float16)x0.z;  b[3]=(_Float16)x0.w;
  b[4]=(_Float16)x1.x;  b[5]=(_Float16)x1.y;  b[6]=(_Float16)x1.z;  b[7]=(_Float16)x1.w;
  b[8]=(_Float16)x2.x;  b[9]=(_Float16)x2.y;  b[10]=(_Float16)x2.z; b[11]=(_Float16)x2.w;
  b[12]=(_Float16)x3.x; b[13]=(_Float16)x3.y; b[14]=(_Float16)x3.z; b[15]=(_Float16)x3.w;
  return b;
}

// ---- two contiguous 8-half LDS runs -> v16h (ds_load_b128 x2)
static __device__ inline v16h lds_pair(const _Float16* p0, const _Float16* p1) {
  v8h lo = *(const v8h*)p0;
  v8h hi = *(const v8h*)p1;
  v16h r;
#pragma unroll
  for (int i = 0; i < 8; ++i) { r[i] = lo[i]; r[8 + i] = hi[i]; }
  return r;
}

// A-panel: whole 20x256 feat tile (2 M-tiles x 8 K-slices), fp32->f16, in VGPRs.
struct APanel { v16h a[2][8]; };

static __device__ inline void load_a_panel(APanel& P, const float* featb, int lane) {
  const bool hi = lane >= 16;
  const int  ln = lane & 15;
#pragma unroll
  for (int mt = 0; mt < 2; ++mt) {
    int t = mt * 16 + ln;
    if (t > TSEQ - 1) t = TSEQ - 1;   // clamp padded rows (no OOB; garbage masked later)
    const float* row = featb + t * CIN;
#pragma unroll
    for (int ks = 0; ks < 8; ++ks) P.a[mt][ks] = a_from_global(row, ks * 32, hi);
  }
}

// ---- one projection: dst = A(20x256) * W^T(256x256), written to LDS as f16.
// MODE 0/2: row-major [t][h]; MODE 1: transposed [h][t] (for the attn@v B-operand).
template <int MODE>
static __device__ void project_gemm(const APanel& P, const float* W,
                                    _Float16* dst, int lane) {
  const bool hi = lane >= 16;
  const int  ln = lane & 15;

#pragma unroll 1
  for (int nt = 0; nt < 16; ++nt) {
    if (nt < 15)  // next weight tile -> global_prefetch_b8 (L2-resident stream)
      __builtin_prefetch(W + ((nt + 1) * 16 + ln) * CIN + (hi ? 16 : 0), 0, 3);
    v8f acc0 = {};
    v8f acc1 = {};
#pragma unroll
    for (int ks = 0; ks < 8; ++ks) {
      v16h b = b_from_global(W, nt * 16 + ln, ks * 32, hi);
      acc0 = WMMA_F16(P.a[0][ks], b, acc0);
      acc1 = WMMA_F16(P.a[1][ks], b, acc1);
    }
    // C/D layout: VGPR r, lanes 0-15 -> M=r, N=lane; lanes 16-31 -> M=r+8, N=lane-16.
#pragma unroll
    for (int r = 0; r < 8; ++r) {
      int row0 = (hi ? 8 : 0) + r;
      int col  = nt * 16 + ln;
      float e0 = acc0[r], e1 = acc1[r];
      if (MODE == 1) {
        dst[col * 32 + row0]         = (_Float16)e0;
        dst[col * 32 + row0 + 16]    = (_Float16)e1;
      } else {
        dst[row0 * 256 + col]        = (_Float16)e0;
        dst[(row0 + 16) * 256 + col] = (_Float16)e1;
      }
    }
  }
}

// Reference's (quirky) banded mask, replicated exactly:
//  rows 0..2 -> cols {0..2}; rows 2..15 additionally -> cols {t, t+1};
//  rows 17..19 -> cols {17..19}; row 16 -> fully masked (softmax -> uniform).
static __device__ inline bool mask_allowed(int t, int s) {
  return (t < 3 && s < 3) ||
         (t >= 17 && s >= 17) ||
         (t >= 2 && t <= 15 && (s == t || s == t + 1));
}

extern "C" __global__ __launch_bounds__(32 * WAVES_PER_BLOCK, 1)
void tca_fused_kernel(const float* __restrict__ feat1, const float* __restrict__ feat2,
                      const float* __restrict__ Wq, const float* __restrict__ Wk,
                      const float* __restrict__ Wv, float* __restrict__ out) {
  extern __shared__ char smem[];
  const int lane = threadIdx.x & 31;
  const int wave = threadIdx.x >> 5;
  const int b    = blockIdx.x * WAVES_PER_BLOCK + wave;
  const bool hi  = lane >= 16;
  const int  ln  = lane & 15;

  char* slot = smem + wave * SLOT_BYTES;
  _Float16* kb   = (_Float16*)(slot + K_OFF);
  _Float16* vtb  = (_Float16*)(slot + VT_OFF);
  _Float16* qb   = (_Float16*)(slot + Q_OFF);
  float*    simb = (float*)   (slot + SIM_OFF);
  _Float16* attb = (_Float16*)(slot + ATT_OFF);

  const float* f1 = feat1 + (size_t)b * TSEQ * CIN;
  const float* f2 = feat2 + (size_t)b * TSEQ * CIN;

  // ---- Phase 1: projections. feat2 panel loaded ONCE, reused for k and v.
  {
    APanel P;
    load_a_panel(P, f2, lane);
    project_gemm<0>(P, Wk, kb,  lane);   // k  row-major
    project_gemm<1>(P, Wv, vtb, lane);   // v  transposed
    load_a_panel(P, f1, lane);           // overwrite panel with feat1
    project_gemm<2>(P, Wq, qb,  lane);   // q  row-major
  }
  LDS_FENCE();

  // ---- Phase 2: sim = q @ k^T / sqrt(HID)   (padded 32x32, K=256)
#pragma unroll
  for (int mt = 0; mt < 2; ++mt) {
#pragma unroll
    for (int nt = 0; nt < 2; ++nt) {
      v8f acc = {};
#pragma unroll
      for (int ks = 0; ks < 8; ++ks) {
        int ac0 = ks * 32 + (hi ? 8 : 0);
        const _Float16* ap = qb + (mt * 16 + ln) * 256 + ac0;
        v16h av = lds_pair(ap, ap + 16);
        int bc0 = ks * 32 + (hi ? 8 : 0);
        const _Float16* bp = kb + (nt * 16 + ln) * 256 + bc0;   // B[c][s] = k[s][c]
        v16h bv = lds_pair(bp, bp + 16);
        acc = WMMA_F16(av, bv, acc);
      }
#pragma unroll
      for (int r = 0; r < 8; ++r) {
        int row = mt * 16 + (hi ? 8 : 0) + r;
        simb[row * 32 + nt * 16 + ln] = acc[r] * 0.0625f;   // 1/sqrt(256)
      }
    }
  }
  LDS_FENCE();

  // ---- Phase 3: masked softmax, one lane per row; f16 attn with zeroed padding
  {
    const int t = lane;
    if (t < TSEQ) {
      float x[TSEQ];
      float mx = -3.0e38f;
#pragma unroll
      for (int s = 0; s < TSEQ; ++s) {
        float v = simb[t * 32 + s];
        v = mask_allowed(t, s) ? v : -1.0e22f;
        x[s] = v;
        mx = fmaxf(mx, v);
      }
      float sum = 0.f;
#pragma unroll
      for (int s = 0; s < TSEQ; ++s) { float e = __expf(x[s] - mx); x[s] = e; sum += e; }
      float inv = 1.0f / sum;
#pragma unroll
      for (int s = 0; s < TSEQ; ++s) attb[t * 32 + s] = (_Float16)(x[s] * inv);
#pragma unroll
      for (int s = TSEQ; s < 32; ++s) attb[t * 32 + s] = (_Float16)0.f;
    } else {
#pragma unroll
      for (int s = 0; s < 32; ++s) attb[t * 32 + s] = (_Float16)0.f;
    }
  }
  LDS_FENCE();

  // ---- Phase 4: out = attn @ v    (K = 32 exactly -> one WMMA per tile)
  v16h aat[2];
#pragma unroll
  for (int mt = 0; mt < 2; ++mt) {
    const _Float16* ap = attb + (mt * 16 + ln) * 32 + (hi ? 8 : 0);
    aat[mt] = lds_pair(ap, ap + 16);
  }
  float* outb = out + (size_t)b * TSEQ * HIDN;
#pragma unroll 1
  for (int nt = 0; nt < 16; ++nt) {
    const _Float16* bp = vtb + (nt * 16 + ln) * 32 + (hi ? 8 : 0);  // B[s][h] = vT[h][s]
    v16h bv = lds_pair(bp, bp + 16);
    v8f acc0 = {};
    v8f acc1 = {};
    acc0 = WMMA_F16(aat[0], bv, acc0);
    acc1 = WMMA_F16(aat[1], bv, acc1);
#pragma unroll
    for (int r = 0; r < 8; ++r) {
      int row0 = (hi ? 8 : 0) + r;
      int col  = nt * 16 + ln;
      outb[row0 * HIDN + col] = acc0[r];            // rows 0..15 always valid
      int row1 = row0 + 16;
      if (row1 < TSEQ) outb[row1 * HIDN + col] = acc1[r];  // rows 16..19 only
    }
  }
}

extern "C" void kernel_launch(void* const* d_in, const int* in_sizes, int n_in,
                              void* d_out, int out_size, void* d_ws, size_t ws_size,
                              hipStream_t stream) {
  (void)in_sizes; (void)n_in; (void)d_ws; (void)ws_size; (void)out_size;
  const float* feat1 = (const float*)d_in[0];
  const float* feat2 = (const float*)d_in[1];
  const float* Wq    = (const float*)d_in[2];
  const float* Wk    = (const float*)d_in[3];
  const float* Wv    = (const float*)d_in[4];
  float*       out   = (float*)d_out;

  // Opt in to >64KB dynamic LDS (not a stream op; safe under graph capture).
  hipFuncSetAttribute((const void*)tca_fused_kernel,
                      hipFuncAttributeMaxDynamicSharedMemorySize, LDS_BYTES);

  dim3 grid(BATCHES / WAVES_PER_BLOCK);
  dim3 block(32 * WAVES_PER_BLOCK);
  tca_fused_kernel<<<grid, block, LDS_BYTES, stream>>>(feat1, feat2, Wq, Wk, Wv, out);
}
